// PerceiverResampler_79018808312130
// MI455X (gfx1250) — matrix operand order repaired and checked by reading, not compiled, not run on hardware
//
#include <hip/hip_runtime.h>
#include <cstdint>
#include <cstddef>

// ---------------------------------------------------------------- types
typedef __attribute__((ext_vector_type(16))) __bf16 v16bf;
typedef __attribute__((ext_vector_type(8)))  __bf16 v8bf;
typedef __attribute__((ext_vector_type(8)))  float  v8f;

#define DEVFN __device__ __forceinline__

// ---------------------------------------------------------------- constants
constexpr int Bb   = 4;
constexpr int Tt   = 8;
constexpr int Ff   = 16;
constexpr int Vv   = 196;
constexpr int DIM  = 768;
constexpr int H    = 12;
constexpr int DH   = 64;
constexpr int HD   = H * DH;       // 768
constexpr int NLAT = 64;
constexpr int DEPTH= 6;
constexpr int DFF  = 4 * DIM;      // 3072
constexpr int NX   = Ff * Vv;      // 3136
constexpr int BT   = Bb * Tt;      // 32
constexpr int ROWS_X = BT * NX;    // 100352
constexpr int NKEY = NX + NLAT;    // 3200
constexpr int MZ   = BT * NLAT;    // 2048
constexpr float SCALE = 0.125f;    // 64^-0.5

// ---------------------------------------------------------------- helpers
DEVFN v16bf cat16(v8bf a, v8bf b) {
  return __builtin_shufflevector(a, b, 0,1,2,3,4,5,6,7,8,9,10,11,12,13,14,15);
}

DEVFN v8f wmma_bf16(v16bf a, v16bf b, v8f c) {
  // D = A(16x32 bf16) * B(32x16 bf16) + C(16x16 f32)
  return __builtin_amdgcn_wmma_f32_16x16x32_bf16(false, a, false, b,
                                                 (short)0, c, false, false);
}

// xor-lane shuffle via ds_swizzle (group-of-32 mode, no index VGPR)
template <int MASK>
DEVFN float swz_xor(float v) {
  int r = __builtin_amdgcn_ds_swizzle(__float_as_int(v), (MASK << 10) | 0x1f);
  return __int_as_float(r);
}

// A fragment: 16x32 tile at (m0,k0) of row-major bf16 matrix, leading dim lda.
// lane l: M = m0+(l&15); lo = K[k0+8*(l>>4) .. +7], hi = K[.. +16 .. +23]
DEVFN v16bf load_a_frag(const __bf16* A, int lda, int m0, int k0, int lane) {
  const __bf16* p = A + (size_t)(m0 + (lane & 15)) * lda + k0 + ((lane >> 4) << 3);
  v8bf lo = *(const v8bf*)p;
  v8bf hi = *(const v8bf*)(p + 16);
  return cat16(lo, hi);
}

// B fragment: 32x16 tile (K x N) taken from PRE-TRANSPOSED Bt (N x K row-major).
// lane l: N = n0+(l&15); lo = K[k0+8*(l>>4) .. +7], hi = K[.. +16 .. +23]
DEVFN v16bf load_b_frag(const __bf16* Bt, int ldb, int n0, int k0, int lane) {
  const __bf16* p = Bt + (size_t)(n0 + (lane & 15)) * ldb + k0 + ((lane >> 4) << 3);
  v8bf lo = *(const v8bf*)p;
  v8bf hi = *(const v8bf*)(p + 16);
  return cat16(lo, hi);
}

// ================================================================ prep kernels

// x + frame_emb + media_emb ; per-row LN stats ; write normalized (no affine) bf16
__global__ __launch_bounds__(256) void prep_x_kernel(const float* __restrict__ x,
                                                     const float* __restrict__ fe,
                                                     const float* __restrict__ me,
                                                     __bf16* __restrict__ xhat) {
  int row = blockIdx.x;                 // [0, ROWS_X)
  int tid = threadIdx.x;
  int rem = row % (Tt * Ff * Vv);
  int t   = rem / (Ff * Vv);
  int f   = (rem % (Ff * Vv)) / Vv;
  const float* xr = x + (size_t)row * DIM;
  const float* fr = fe + (size_t)f * DIM;
  const float* mr = me + (size_t)t * DIM;   // media_emb (MAXT,1,DIM)

  float vals[3];
  float s = 0.f, s2 = 0.f;
  #pragma unroll
  for (int j = 0; j < 3; ++j) {
    int d = tid + j * 256;
    float v = xr[d] + fr[d] + mr[d];
    vals[j] = v; s += v; s2 += v * v;
  }
  __shared__ float rs[256], rq[256];
  rs[tid] = s; rq[tid] = s2; __syncthreads();
  for (int st = 128; st > 0; st >>= 1) {
    if (tid < st) { rs[tid] += rs[tid + st]; rq[tid] += rq[tid + st]; }
    __syncthreads();
  }
  float mu  = rs[0] * (1.f / DIM);
  float var = rq[0] * (1.f / DIM) - mu * mu;
  float rstd = rsqrtf(var + 1e-5f);
  __bf16* orow = xhat + (size_t)row * DIM;
  #pragma unroll
  for (int j = 0; j < 3; ++j) {
    int d = tid + j * 256;
    orow[d] = (__bf16)((vals[j] - mu) * rstd);
  }
}

// LN over f32 rows with affine; output bf16 or f32
template <bool OUT_F32>
__global__ __launch_bounds__(256) void ln_rows_kernel(const float* __restrict__ in,
                                                      const float* __restrict__ g,
                                                      const float* __restrict__ b,
                                                      void* __restrict__ out) {
  int row = blockIdx.x;
  int tid = threadIdx.x;
  const float* xr = in + (size_t)row * DIM;
  float vals[3];
  float s = 0.f, s2 = 0.f;
  #pragma unroll
  for (int j = 0; j < 3; ++j) {
    int d = tid + j * 256;
    float v = xr[d];
    vals[j] = v; s += v; s2 += v * v;
  }
  __shared__ float rs[256], rq[256];
  rs[tid] = s; rq[tid] = s2; __syncthreads();
  for (int st = 128; st > 0; st >>= 1) {
    if (tid < st) { rs[tid] += rs[tid + st]; rq[tid] += rq[tid + st]; }
    __syncthreads();
  }
  float mu  = rs[0] * (1.f / DIM);
  float var = rq[0] * (1.f / DIM) - mu * mu;
  float rstd = rsqrtf(var + 1e-5f);
  #pragma unroll
  for (int j = 0; j < 3; ++j) {
    int d = tid + j * 256;
    float v = (vals[j] - mu) * rstd * g[d] + b[d];
    if (OUT_F32) ((float*)out)[(size_t)row * DIM + d] = v;
    else         ((__bf16*)out)[(size_t)row * DIM + d] = (__bf16)v;
  }
}

__global__ void init_z_kernel(const float* __restrict__ lat, float* __restrict__ z) {
  int idx = blockIdx.x * 256 + threadIdx.x;
  if (idx >= MZ * DIM) return;
  int d = idx % DIM;
  int i = (idx / DIM) % NLAT;
  z[idx] = lat[i * DIM + d];
}

// Wt[n][k] = (g ? g[k] : 1) * W[k][n], f32 -> bf16, pre-transposed for B operand
__global__ void transpose_w_kernel(const float* __restrict__ W,
                                   const float* __restrict__ g,
                                   int K, int N, __bf16* __restrict__ Wt) {
  int idx = blockIdx.x * 256 + threadIdx.x;
  if (idx >= K * N) return;
  int n = idx % N, k = idx / N;
  float s = g ? g[k] : 1.f;
  Wt[(size_t)n * K + k] = (__bf16)(s * W[idx]);
}

// bias[n] = sum_k bvec[k] * W[k][n]
__global__ void bias_row_kernel(const float* __restrict__ bvec,
                                const float* __restrict__ W,
                                int K, int N, float* __restrict__ bias) {
  int n = blockIdx.x * blockDim.x + threadIdx.x;
  if (n >= N) return;
  float s = 0.f;
  for (int k = 0; k < K; ++k) s += bvec[k] * W[(size_t)k * N + n];
  bias[n] = s;
}

// ================================================================ generic WMMA GEMM
// C(MxN) = A(MxK bf16, row-major) * Bt^T (Bt is NxK bf16 row-major) [+ bias row]
// Block tile 128x128, 8 waves (4Mx2N), wave tile 32x64 (2x4 fragments).
struct GemmP {
  const __bf16* A;  int lda;
  const __bf16* Bt; int ldb;
  int M, N, K;
  const float* bias;          // nullable (added per output column)
  // epilogue payloads
  float*  zres;               // EPI 2: residual accumulate (MxN f32)
  __bf16* outb;               // EPI 1/3: bf16 output (MxN)
  __bf16* kbuf;               // EPI 0
  __bf16* vtbuf;              // EPI 0
  int rows_per_bt;            // EPI 0/1
  int key_base;               // EPI 0
  const unsigned char* mask;  // EPI 1
  float qscale;               // EPI 1
};

template <int EPI>
__global__ __launch_bounds__(256) void gemm_kernel(GemmP p) {
  int lane = threadIdx.x & 31;
  int wid  = threadIdx.x >> 5;
  int wm   = wid & 3;             // 4 waves along M
  int wn   = wid >> 2;            // 2 waves along N
  int m0   = blockIdx.y * 128 + wm * 32;
  int n0   = blockIdx.x * 128 + wn * 64;

  v8f acc[2][4] = {};
  for (int k0 = 0; k0 < p.K; k0 += 32) {
    v16bf a0 = load_a_frag(p.A, p.lda, m0,      k0, lane);
    v16bf a1 = load_a_frag(p.A, p.lda, m0 + 16, k0, lane);
    v16bf b[4];
    #pragma unroll
    for (int j = 0; j < 4; ++j) b[j] = load_b_frag(p.Bt, p.ldb, n0 + j * 16, k0, lane);
    #pragma unroll
    for (int j = 0; j < 4; ++j) {
      acc[0][j] = wmma_bf16(a0, b[j], acc[0][j]);
      acc[1][j] = wmma_bf16(a1, b[j], acc[1][j]);
    }
  }

  int lh = lane & 15, lg = lane >> 4;
  #pragma unroll
  for (int i = 0; i < 2; ++i) {
    #pragma unroll
    for (int j = 0; j < 4; ++j) {
      int nb = n0 + j * 16;
      int n  = nb + lh;
      int mb = m0 + i * 16 + lg * 8;      // rows mb..mb+7 (contiguous)
      float bv = p.bias ? p.bias[n] : 0.f;
      if constexpr (EPI == 0) {           // KV: split K / V(transposed, packed store)
        int bt   = mb / p.rows_per_bt;    // uniform over r (rows_per_bt % 8 == 0)
        int key0 = p.key_base + mb % p.rows_per_bt;
        if (nb < HD) {
          #pragma unroll
          for (int r = 0; r < 8; ++r)
            p.kbuf[((size_t)bt * NKEY + key0 + r) * HD + n] = (__bf16)(acc[i][j][r] + bv);
        } else {
          v8bf pk;
          #pragma unroll
          for (int r = 0; r < 8; ++r) pk[r] = (__bf16)(acc[i][j][r] + bv);
          *(v8bf*)&p.vtbuf[((size_t)bt * HD + (n - HD)) * NKEY + key0] = pk;
        }
      } else if constexpr (EPI == 1) {    // Q: * SCALE * mask -> bf16
        int bt = mb / p.rows_per_bt;
        float s = p.qscale * (p.mask[bt] ? 1.f : 0.f);
        #pragma unroll
        for (int r = 0; r < 8; ++r)
          p.outb[(size_t)(mb + r) * p.N + n] = (__bf16)((acc[i][j][r] + bv) * s);
      } else if constexpr (EPI == 2) {    // residual accumulate into f32
        #pragma unroll
        for (int r = 0; r < 8; ++r)
          p.zres[(size_t)(mb + r) * p.N + n] += acc[i][j][r] + bv;
      } else {                            // EPI == 3: exact GELU -> bf16
        #pragma unroll
        for (int r = 0; r < 8; ++r) {
          float v = acc[i][j][r] + bv;
          float gl = 0.5f * v * (1.f + erff(v * 0.70710678118654752440f));
          p.outb[(size_t)(mb + r) * p.N + n] = (__bf16)gl;
        }
      }
    }
  }
}

// ================================================================ flash attention
// grid (BT, H); block = 128 threads (4 waves); wave handles 16 query rows.
// 64 keys per iteration.  Row sums come from a "ones-column" WMMA (no shuffles);
// row max uses one 4-step ds_swizzle reduction per row per 64 keys.
// q pre-scaled by SCALE*mask.  K: [bt][key][h*64+d] bf16.  V^T: [bt][h*64+d][key] bf16.
__global__ __launch_bounds__(128) void attn_kernel(const __bf16* __restrict__ qb,
                                                   const __bf16* __restrict__ kb,
                                                   const __bf16* __restrict__ vtb,
                                                   const unsigned char* __restrict__ mask,
                                                   __bf16* __restrict__ aout) {
  int bt = blockIdx.x, h = blockIdx.y;
  int wid = threadIdx.x >> 5, lane = threadIdx.x & 31;
  int lh = lane & 15, lg = lane >> 4;
  int m0 = wid * 16;

  if (!mask[bt]) {   // whole (b,t) masked out -> output 0 (uniform branch)
    #pragma unroll
    for (int r = 0; r < 8; ++r) {
      int m = m0 + lg * 8 + r;
      __bf16* p = aout + ((size_t)bt * NLAT + m) * HD + h * DH;
      #pragma unroll
      for (int nt = 0; nt < 4; ++nt) p[nt * 16 + lh] = (__bf16)0.f;
    }
    return;
  }

  __shared__ __attribute__((aligned(16))) __bf16 psh[4][16 * 64];

  // load Q fragments (16 rows x 64 dh = 2 K-chunks)
  v16bf qa[2];
  {
    const __bf16* p = qb + ((size_t)bt * NLAT + m0 + lh) * HD + h * DH;
    #pragma unroll
    for (int c = 0; c < 2; ++c) {
      int kk = c * 32 + lg * 8;
      qa[c] = cat16(*(const v8bf*)(p + kk), *(const v8bf*)(p + kk + 16));
    }
  }

  v16bf ones;
  #pragma unroll
  for (int i = 0; i < 16; ++i) ones[i] = (__bf16)1.f;

  float rmax[8];
  #pragma unroll
  for (int r = 0; r < 8; ++r) rmax[r] = -INFINITY;
  v8f o[4] = {};
  v8f sacc = {};                        // replicated row sums via ones-column WMMA

  const __bf16* kbase = kb  + (size_t)bt * NKEY * HD + h * DH;
  const __bf16* vbase = vtb + ((size_t)bt * HD + h * DH) * NKEY;

  for (int j0 = 0; j0 < NKEY; j0 += 64) {
    // sim tiles: s[t] = keys j0+16t .. j0+16t+15
    v8f s[4] = {};
    #pragma unroll
    for (int t = 0; t < 4; ++t) {
      #pragma unroll
      for (int c = 0; c < 2; ++c) {
        const __bf16* p = kbase + (size_t)(j0 + t * 16 + lh) * HD + c * 32 + lg * 8;
        s[t] = wmma_bf16(qa[c], cat16(*(const v8bf*)p, *(const v8bf*)(p + 16)), s[t]);
      }
    }
    // online max update (one reduction per row per 64 keys)
    float cs[8], pv[4][8];
    #pragma unroll
    for (int r = 0; r < 8; ++r) {
      float v = fmaxf(fmaxf(s[0][r], s[1][r]), fmaxf(s[2][r], s[3][r]));
      v = fmaxf(v, swz_xor<1>(v));
      v = fmaxf(v, swz_xor<2>(v));
      v = fmaxf(v, swz_xor<4>(v));
      v = fmaxf(v, swz_xor<8>(v));
      float nm = fmaxf(rmax[r], v);
      cs[r]   = __expf(rmax[r] - nm);
      rmax[r] = nm;
      #pragma unroll
      for (int t = 0; t < 4; ++t) pv[t][r] = __expf(s[t][r] - nm);
    }
    // correct accumulators
    #pragma unroll
    for (int r = 0; r < 8; ++r) {
      sacc[r] *= cs[r];
      #pragma unroll
      for (int nt = 0; nt < 4; ++nt) o[nt][r] *= cs[r];
    }
    // P (C-layout) -> LDS -> A-layout fragments
    #pragma unroll
    for (int t = 0; t < 4; ++t)
      #pragma unroll
      for (int r = 0; r < 8; ++r)
        psh[wid][(lg * 8 + r) * 64 + t * 16 + lh] = (__bf16)pv[t][r];
    asm volatile("s_wait_dscnt 0x0" ::: "memory");
    #pragma unroll
    for (int c = 0; c < 2; ++c) {
      const __bf16* sp = &psh[wid][lh * 64 + c * 32 + lg * 8];
      v16bf pa = cat16(*(const v8bf*)sp, *(const v8bf*)(sp + 16));
      sacc = wmma_bf16(pa, ones, sacc);   // row sums, replicated across lanes
      #pragma unroll
      for (int nt = 0; nt < 4; ++nt) {
        const __bf16* vp = vbase + (size_t)(nt * 16 + lh) * NKEY + j0 + c * 32 + lg * 8;
        o[nt] = wmma_bf16(pa, cat16(*(const v8bf*)vp, *(const v8bf*)(vp + 16)), o[nt]);
      }
    }
  }

  #pragma unroll
  for (int nt = 0; nt < 4; ++nt)
    #pragma unroll
    for (int r = 0; r < 8; ++r) {
      int m = m0 + lg * 8 + r;
      float v = o[nt][r] / sacc[r];
      aout[((size_t)bt * NLAT + m) * HD + h * DH + nt * 16 + lh] = (__bf16)v;
    }
}

// ================================================================ host side
static void launch_gemm(int epi, const GemmP& p, hipStream_t s) {
  dim3 grid(p.N / 128, p.M / 128);
  switch (epi) {
    case 0: gemm_kernel<0><<<grid, 256, 0, s>>>(p); break;
    case 1: gemm_kernel<1><<<grid, 256, 0, s>>>(p); break;
    case 2: gemm_kernel<2><<<grid, 256, 0, s>>>(p); break;
    default: gemm_kernel<3><<<grid, 256, 0, s>>>(p); break;
  }
}

extern "C" void kernel_launch(void* const* d_in, const int* in_sizes, int n_in,
                              void* d_out, int out_size, void* d_ws, size_t ws_size,
                              hipStream_t stream) {
  (void)in_sizes; (void)n_in; (void)out_size; (void)ws_size;
  const float* x    = (const float*)d_in[0];
  const unsigned char* mask = (const unsigned char*)d_in[1];   // bool (b,T)
  const float* lat  = (const float*)d_in[2];
  const float* fe   = (const float*)d_in[3];
  const float* me   = (const float*)d_in[4];
  const float* q_w  = (const float*)d_in[5];
  const float* kv_w = (const float*)d_in[6];
  const float* o_w  = (const float*)d_in[7];
  const float* lnm_g = (const float*)d_in[8];
  const float* lnm_b = (const float*)d_in[9];
  const float* lnl_g = (const float*)d_in[10];
  const float* lnl_b = (const float*)d_in[11];
  const float* ffln_g = (const float*)d_in[12];
  const float* ffln_b = (const float*)d_in[13];
  const float* ff_w1 = (const float*)d_in[14];
  const float* ff_w2 = (const float*)d_in[15];
  const float* fin_g = (const float*)d_in[16];
  const float* fin_b = (const float*)d_in[17];
  float* out = (float*)d_out;

  size_t off = 0;
  auto alloc = [&](size_t bytes) -> char* {
    char* p = (char*)d_ws + off;
    off += (bytes + 255) & ~(size_t)255;
    return p;
  };
  __bf16* xhat  = (__bf16*)alloc((size_t)ROWS_X * DIM * 2);
  __bf16* kbuf  = (__bf16*)alloc((size_t)BT * NKEY * HD * 2);
  __bf16* vtbuf = (__bf16*)alloc((size_t)BT * HD * NKEY * 2);
  float*  z     = (float*) alloc((size_t)MZ * DIM * 4);
  __bf16* zn    = (__bf16*)alloc((size_t)MZ * DIM * 2);
  __bf16* qbuf  = (__bf16*)alloc((size_t)MZ * HD * 2);
  __bf16* aout  = (__bf16*)alloc((size_t)MZ * HD * 2);
  __bf16* hn    = (__bf16*)alloc((size_t)MZ * DIM * 2);
  __bf16* hbuf  = (__bf16*)alloc((size_t)MZ * DFF * 2);
  __bf16* wkvm  = (__bf16*)alloc((size_t)2 * HD * DIM * 2);  // (1536 x 768) Bt
  __bf16* wkvz  = (__bf16*)alloc((size_t)2 * HD * DIM * 2);
  __bf16* wq    = (__bf16*)alloc((size_t)HD * DIM * 2);
  __bf16* wo    = (__bf16*)alloc((size_t)DIM * HD * 2);
  __bf16* wf1   = (__bf16*)alloc((size_t)DFF * DIM * 2);
  __bf16* wf2   = (__bf16*)alloc((size_t)DIM * DFF * 2);
  float*  biaskv= (float*) alloc((size_t)2 * HD * 4);

  // ---- one-time prep
  prep_x_kernel<<<ROWS_X, 256, 0, stream>>>(x, fe, me, xhat);
  init_z_kernel<<<(MZ * DIM + 255) / 256, 256, 0, stream>>>(lat, z);

  for (int i = 0; i < DEPTH; ++i) {
    const float* kvw_i = kv_w + (size_t)i * DIM * 2 * HD;
    const float* qw_i  = q_w  + (size_t)i * DIM * HD;
    const float* ow_i  = o_w  + (size_t)i * HD * DIM;
    const float* f1_i  = ff_w1 + (size_t)i * DIM * DFF;
    const float* f2_i  = ff_w2 + (size_t)i * DFF * DIM;

    // weight fold/transpose (f32 -> bf16, N x K layout for B operands)
    transpose_w_kernel<<<(DIM * 2 * HD + 255) / 256, 256, 0, stream>>>(
        kvw_i, lnm_g + i * DIM, DIM, 2 * HD, wkvm);
    bias_row_kernel<<<(2 * HD + 255) / 256, 256, 0, stream>>>(
        lnm_b + i * DIM, kvw_i, DIM, 2 * HD, biaskv);
    transpose_w_kernel<<<(DIM * 2 * HD + 255) / 256, 256, 0, stream>>>(
        kvw_i, nullptr, DIM, 2 * HD, wkvz);
    transpose_w_kernel<<<(DIM * HD + 255) / 256, 256, 0, stream>>>(
        qw_i, nullptr, DIM, HD, wq);
    transpose_w_kernel<<<(HD * DIM + 255) / 256, 256, 0, stream>>>(
        ow_i, nullptr, HD, DIM, wo);
    transpose_w_kernel<<<(DIM * DFF + 255) / 256, 256, 0, stream>>>(
        f1_i, nullptr, DIM, DFF, wf1);
    transpose_w_kernel<<<(DFF * DIM + 255) / 256, 256, 0, stream>>>(
        f2_i, nullptr, DFF, DIM, wf2);

    // zn = LN_l(z)  (affine applied)
    ln_rows_kernel<false><<<MZ, 256, 0, stream>>>(z, lnl_g + i * DIM, lnl_b + i * DIM, zn);

    // KV, x-part: xhat @ (g_m . kvw) + b_m @ kvw
    GemmP pkx{};
    pkx.A = xhat; pkx.lda = DIM; pkx.Bt = wkvm; pkx.ldb = DIM;
    pkx.M = ROWS_X; pkx.N = 2 * HD; pkx.K = DIM; pkx.bias = biaskv;
    pkx.kbuf = kbuf; pkx.vtbuf = vtbuf; pkx.rows_per_bt = NX; pkx.key_base = 0;
    launch_gemm(0, pkx, stream);

    // KV, z-part: zn @ kvw
    GemmP pkz = pkx;
    pkz.A = zn; pkz.M = MZ; pkz.Bt = wkvz; pkz.bias = nullptr;
    pkz.rows_per_bt = NLAT; pkz.key_base = NX;
    launch_gemm(0, pkz, stream);

    // Q = (zn @ qw) * SCALE * mask
    GemmP pq{};
    pq.A = zn; pq.lda = DIM; pq.Bt = wq; pq.ldb = DIM;
    pq.M = MZ; pq.N = HD; pq.K = DIM;
    pq.outb = qbuf; pq.rows_per_bt = NLAT; pq.mask = mask; pq.qscale = SCALE;
    launch_gemm(1, pq, stream);

    // attention
    attn_kernel<<<dim3(BT, H), 128, 0, stream>>>(qbuf, kbuf, vtbuf, mask, aout);

    // z += aout @ ow
    GemmP po{};
    po.A = aout; po.lda = HD; po.Bt = wo; po.ldb = HD;
    po.M = MZ; po.N = DIM; po.K = HD; po.zres = z;
    launch_gemm(2, po, stream);

    // hn = LN_ff(z)
    ln_rows_kernel<false><<<MZ, 256, 0, stream>>>(z, ffln_g + i * DIM, ffln_b + i * DIM, hn);

    // hbuf = gelu(hn @ w1)
    GemmP pf1{};
    pf1.A = hn; pf1.lda = DIM; pf1.Bt = wf1; pf1.ldb = DIM;
    pf1.M = MZ; pf1.N = DFF; pf1.K = DIM; pf1.outb = hbuf;
    launch_gemm(3, pf1, stream);

    // z += hbuf @ w2
    GemmP pf2{};
    pf2.A = hbuf; pf2.lda = DFF; pf2.Bt = wf2; pf2.ldb = DFF;
    pf2.M = MZ; pf2.N = DIM; pf2.K = DFF; pf2.zres = z;
    launch_gemm(2, pf2, stream);
  }

  // final LN -> f32 output
  ln_rows_kernel<true><<<MZ, 256, 0, stream>>>(z, fin_g, fin_b, out);
}
